// MultiheadAttention_80573586472989
// MI455X (gfx1250) — compile-verified, matrix operand order checked by reading, not run on hardware
//
#include <hip/hip_runtime.h>
#include <hip/hip_bf16.h>
#include <cstdint>

typedef __attribute__((ext_vector_type(16))) _Float16 v16h;
typedef __attribute__((ext_vector_type(8)))  _Float16 v8h;
typedef __attribute__((ext_vector_type(8)))  float    v8f;
typedef __attribute__((ext_vector_type(2)))  _Float16 v2h;

// gfx1250 async memory->LDS path (ASYNCcnt), if this toolchain exposes it.
#if defined(__gfx1250__) && \
    __has_builtin(__builtin_amdgcn_global_load_async_to_lds_b128) && \
    __has_builtin(__builtin_amdgcn_s_wait_asynccnt)
#define USE_ASYNC_LDS 1
#define AS1 __attribute__((address_space(1)))
#define AS3 __attribute__((address_space(3)))
typedef __attribute__((ext_vector_type(4))) int v4i;
#else
#define USE_ASYNC_LDS 0
#endif

// ---------------------------------------------------------------------------
// Fragment loader for V_WMMA_F32_16X16X32_F16 A/B operands (wave32).
// 16-bit A-matrix 16x32 layout (ISA 7.12.2): lane L (0-15) holds row M=L,
// halves hold K pairs {0,1},{2,3},{4,5},{6,7},{16,17},... ; lanes 16-31 hold
// the same rows with K offset +8. B (32x16) mirrors with N for M. Works for
// global or LDS pointers (generic addressing); coalesces to 2x b128 per frag.
// ---------------------------------------------------------------------------
__device__ __forceinline__ v16h load_frag(const _Float16* __restrict__ base,
                                          int ld, int row, int kbase, int lane) {
  const _Float16* rp = base + (size_t)(row + (lane & 15)) * ld
                            + kbase + ((lane >> 4) << 3);
  v16h out;
#pragma unroll
  for (int t = 0; t < 8; ++t) {
    int kk = 2 * t + ((t >= 4) ? 8 : 0);
    v2h p = *(const v2h*)(rp + kk);
    out[2 * t]     = p[0];
    out[2 * t + 1] = p[1];
  }
  return out;
}

__device__ __forceinline__ v8f wmma32(v16h a, v16h b, v8f c) {
  return __builtin_amdgcn_wmma_f32_16x16x32_f16(
      false, a, false, b, (short)0, c, false, false);
}

// ---------------------------------------------------------------------------
// fp32 -> f16 convert / transpose-convert
// ---------------------------------------------------------------------------
__global__ void cvt_f16(const float* __restrict__ in, _Float16* __restrict__ out, int n) {
  int i = blockIdx.x * blockDim.x + threadIdx.x;
  if (i < n) out[i] = (_Float16)in[i];
}

// out[n*K + k] = in[k*N + n]  (weights -> [N][K] so B frags stream along K)
__global__ void cvtT_f16(const float* __restrict__ in, _Float16* __restrict__ out,
                         int N, int K) {
  int i = blockIdx.x * blockDim.x + threadIdx.x;
  if (i < N * K) {
    int n = i / K;
    int k = i - n * K;
    out[i] = (_Float16)in[(size_t)k * N + n];
  }
}

// ---------------------------------------------------------------------------
// 32x64-per-wave GEMM core, K=1024, both operands K-contiguous with ld=1024.
// Two named fragment sets (A/B) alternate roles each half-iteration: loads
// for one set issue while the other's 8 WMMAs execute — no register copies,
// so the scheduler overlaps loadcnt waits with WMMA issue.
// ---------------------------------------------------------------------------
__device__ __forceinline__ void load_set(const _Float16* __restrict__ A,
                                         const _Float16* __restrict__ Bt,
                                         int mrow, int ncol, int k, int lane,
                                         v16h& a0, v16h& a1, v16h b[4]) {
  a0 = load_frag(A, 1024, mrow,      k, lane);
  a1 = load_frag(A, 1024, mrow + 16, k, lane);
#pragma unroll
  for (int g = 0; g < 4; ++g) b[g] = load_frag(Bt, 1024, ncol + 16 * g, k, lane);
}

__device__ __forceinline__ void mma_step(v8f acc[2][4], const v16h& a0,
                                         const v16h& a1, const v16h b[4]) {
#pragma unroll
  for (int g = 0; g < 4; ++g) {
    acc[0][g] = wmma32(a0, b[g], acc[0][g]);
    acc[1][g] = wmma32(a1, b[g], acc[1][g]);
  }
}

__device__ __forceinline__ void gemm_core(const _Float16* __restrict__ A,
                                          const _Float16* __restrict__ Bt,
                                          int mrow, int ncol, int lane,
                                          v8f acc[2][4]) {
  v8f z = {};
#pragma unroll
  for (int i = 0; i < 2; ++i)
#pragma unroll
    for (int g = 0; g < 4; ++g) acc[i][g] = z;

  v16h aA0, aA1, bA[4], aB0, aB1, bB[4];
  load_set(A, Bt, mrow, ncol, 0, lane, aA0, aA1, bA);
  for (int k = 0; k < 1024; k += 64) {
    load_set(A, Bt, mrow, ncol, k + 32, lane, aB0, aB1, bB);
    mma_step(acc, aA0, aA1, bA);
    load_set(A, Bt, mrow, ncol, (k + 64) & 1023, lane, aA0, aA1, bA); // wraps once
    mma_step(acc, aB0, aB1, bB);
  }
}

// ---------------------------------------------------------------------------
// QKV GEMM: [4096,1024] x [1024,3072] (+bias). Epilogue routes q/k/v, folds
// softmax scale into q,k, stores V transposed per head [d][s].
// ---------------------------------------------------------------------------
__global__ void __launch_bounds__(256)
gemm_qkv(const _Float16* __restrict__ xh, const _Float16* __restrict__ wt,
         const float* __restrict__ bias,
         _Float16* __restrict__ qh, _Float16* __restrict__ kh,
         _Float16* __restrict__ vth) {
  const int lane = threadIdx.x & 31;
  const int wid  = blockIdx.x * (blockDim.x >> 5) + (threadIdx.x >> 5);
  const int NG   = 3072 / 64;
  const int mt   = wid / NG;
  const int ng   = wid - mt * NG;
  const int mrow = mt * 32;
  const int ncol = ng * 64;

  v8f acc[2][4];
  gemm_core(xh, wt, mrow, ncol, lane, acc);

  const float scale = 0.35355339059327373f;   // 64^-0.25
  const int nlo = lane & 15;
  const int mhi = (lane >> 4) << 3;
#pragma unroll
  for (int mi = 0; mi < 2; ++mi) {
#pragma unroll
    for (int g = 0; g < 4; ++g) {
#pragma unroll
      for (int r = 0; r < 8; ++r) {
        int m = mrow + mi * 16 + r + mhi;
        int n = ncol + 16 * g + nlo;
        float v = acc[mi][g][r] + bias[n];
        int head = n / 192, within = n - head * 192;
        int which = within >> 6, c = within & 63;
        int b_ = m >> 11, s = m & 2047;
        int bh = b_ * 16 + head;
        if (which == 0)
          qh[((size_t)bh * 2048 + s) * 64 + c] = (_Float16)(v * scale);
        else if (which == 1)
          kh[((size_t)bh * 2048 + s) * 64 + c] = (_Float16)(v * scale);
        else
          vth[((size_t)bh * 64 + c) * 2048 + s] = (_Float16)v;
      }
    }
  }
}

// ---------------------------------------------------------------------------
// Flash attention, block-cooperative with DOUBLE-BUFFERED LDS staging:
// a 256-thread block owns one (b,h) and 128 queries (wave w -> 16 rows).
// While all 8 waves compute on chunk j from buffer p, chunk j+32 streams
// into buffer p^1 — via GLOBAL_LOAD_ASYNC_TO_LDS_B128 (ASYNCcnt, no VGPR
// round-trip) when available, else plain b128 copies. One barrier per chunk.
// ---------------------------------------------------------------------------
__global__ void __launch_bounds__(256)
attn_fwd(const _Float16* __restrict__ qh, const _Float16* __restrict__ kh,
         const _Float16* __restrict__ vth, _Float16* __restrict__ attnh) {
  __shared__ __align__(16) _Float16 kbuf[2][32 * 64];  // keys x d   (2x4 KB)
  __shared__ __align__(16) _Float16 vbuf[2][64 * 32];  // d x keys   (2x4 KB)
  __shared__ __align__(16) _Float16 plds[8 * 512];     // per-wave P (8 KB)

  const int tid  = threadIdx.x;
  const int w    = tid >> 5;
  const int lane = tid & 31;
  const int bh   = blockIdx.x >> 4;                    // 0..31 (b*16+h)
  const int qb   = blockIdx.x & 15;

  const _Float16* q    = qh  + (size_t)bh * 2048 * 64;
  const _Float16* kmat = kh  + (size_t)bh * 2048 * 64;
  const _Float16* vt   = vth + (size_t)bh * 64 * 2048;
  _Float16* pl = plds + w * 512;

  const int qrow = qb * 128 + w * 16;
  v16h qa0 = load_frag(q, 64, qrow, 0,  lane);
  v16h qa1 = load_frag(q, 64, qrow, 32, lane);

  v8f z = {};
  v8f o[4] = {z, z, z, z};
  float mrun[8], lrun[8];
#pragma unroll
  for (int r = 0; r < 8; ++r) { mrun[r] = -3.0e38f; lrun[r] = 0.0f; }

  const int nlo = lane & 15;
  const int mhi = (lane >> 4) << 3;
  // staging: one 16-byte segment per thread per buffer
  const int krow = tid >> 3, kseg = (tid & 7) * 8;     // 32 rows x 64 halves
  const int vrow = tid >> 2, vseg = (tid & 3) * 8;     // 64 rows x 32 halves

  // stage chunk starting at key j into buffer `buf`
  auto stage = [&](int buf, int j) {
    const _Float16* ksrc = &kmat[(size_t)(j + krow) * 64 + kseg];
    const _Float16* vsrc = &vt[(size_t)vrow * 2048 + j + vseg];
    _Float16* kdst = &kbuf[buf][krow * 64 + kseg];
    _Float16* vdst = &vbuf[buf][vrow * 32 + vseg];
#if USE_ASYNC_LDS
    __builtin_amdgcn_global_load_async_to_lds_b128(
        (AS1 v4i*)(uintptr_t)ksrc,
        (AS3 v4i*)(uint32_t)(uintptr_t)kdst, 0, 0);
    __builtin_amdgcn_global_load_async_to_lds_b128(
        (AS1 v4i*)(uintptr_t)vsrc,
        (AS3 v4i*)(uint32_t)(uintptr_t)vdst, 0, 0);
#else
    *(v8h*)kdst = *(const v8h*)ksrc;
    *(v8h*)vdst = *(const v8h*)vsrc;
#endif
  };

  stage(0, 0);
#if USE_ASYNC_LDS
  __builtin_amdgcn_s_wait_asynccnt(0);
#endif
  __syncthreads();

  int p = 0;
  for (int j = 0; j < 2048; j += 32) {
    stage(p ^ 1, (j + 32) & 2047);                     // overlap next chunk

    const _Float16* kb = kbuf[p];
    const _Float16* vb = vbuf[p];

    // S tiles: keys 0..15 and 16..31 of the chunk, d=64 in 2 k-steps
    v8f s0 = z, s1 = z;
    {
      v16h b = load_frag(kb, 64, 0, 0, lane);
      s0 = wmma32(qa0, b, s0);
      b = load_frag(kb, 64, 0, 32, lane);
      s0 = wmma32(qa1, b, s0);
      b = load_frag(kb, 64, 16, 0, lane);
      s1 = wmma32(qa0, b, s1);
      b = load_frag(kb, 64, 16, 32, lane);
      s1 = wmma32(qa1, b, s1);
    }

    // online softmax; C-layout rows live in 16-lane halves
#pragma unroll
    for (int r = 0; r < 8; ++r) {
      float mx = fmaxf(s0[r], s1[r]);
      mx = fmaxf(mx, __shfl_xor(mx, 1, 32));
      mx = fmaxf(mx, __shfl_xor(mx, 2, 32));
      mx = fmaxf(mx, __shfl_xor(mx, 4, 32));
      mx = fmaxf(mx, __shfl_xor(mx, 8, 32));
      float mnew = fmaxf(mrun[r], mx);
      float p0 = __expf(s0[r] - mnew);
      float p1 = __expf(s1[r] - mnew);
      float rs = p0 + p1;
      rs += __shfl_xor(rs, 1, 32);
      rs += __shfl_xor(rs, 2, 32);
      rs += __shfl_xor(rs, 4, 32);
      rs += __shfl_xor(rs, 8, 32);
      float alpha = __expf(mrun[r] - mnew);
      lrun[r] = alpha * lrun[r] + rs;
      mrun[r] = mnew;
#pragma unroll
      for (int g = 0; g < 4; ++g) o[g][r] *= alpha;
      int m = r + mhi;
      pl[m * 32 + nlo]      = (_Float16)p0;
      pl[m * 32 + 16 + nlo] = (_Float16)p1;
    }

    // O += P @ V
    v16h pa = load_frag(pl, 32, 0, 0, lane);
#pragma unroll
    for (int g = 0; g < 4; ++g) {
      v16h vf = load_frag(vb, 32, 16 * g, 0, lane);
      o[g] = wmma32(pa, vf, o[g]);
    }

#if USE_ASYNC_LDS
    __builtin_amdgcn_s_wait_asynccnt(0);
#endif
    __syncthreads();                                   // next buffer ready
    p ^= 1;
  }

  // normalize + store f16 row-major [4096 x 1024]
  const int b_ = bh >> 4, h = bh & 15;
#pragma unroll
  for (int g = 0; g < 4; ++g) {
#pragma unroll
    for (int r = 0; r < 8; ++r) {
      int m = qrow + r + mhi;
      float val = o[g][r] / lrun[r];
      attnh[((size_t)(b_ * 2048 + m)) * 1024 + h * 64 + g * 16 + nlo] =
          (_Float16)val;
    }
  }
}

// ---------------------------------------------------------------------------
// Output projection: [4096,1024] x [1024,1024] (+bias) -> fp32 d_out
// ---------------------------------------------------------------------------
__global__ void __launch_bounds__(256)
gemm_proj(const _Float16* __restrict__ ah, const _Float16* __restrict__ wt,
          const float* __restrict__ bias, float* __restrict__ out) {
  const int lane = threadIdx.x & 31;
  const int wid  = blockIdx.x * (blockDim.x >> 5) + (threadIdx.x >> 5);
  const int NG   = 1024 / 64;
  const int mt   = wid / NG;
  const int ng   = wid - mt * NG;
  const int mrow = mt * 32;
  const int ncol = ng * 64;

  v8f acc[2][4];
  gemm_core(ah, wt, mrow, ncol, lane, acc);

  const int nlo = lane & 15;
  const int mhi = (lane >> 4) << 3;
#pragma unroll
  for (int mi = 0; mi < 2; ++mi) {
#pragma unroll
    for (int g = 0; g < 4; ++g) {
#pragma unroll
      for (int r = 0; r < 8; ++r) {
        int m = mrow + mi * 16 + r + mhi;
        int n = ncol + 16 * g + nlo;
        out[(size_t)m * 1024 + n] = acc[mi][g][r] + bias[n];
      }
    }
  }
}

// ---------------------------------------------------------------------------
extern "C" void kernel_launch(void* const* d_in, const int* in_sizes, int n_in,
                              void* d_out, int out_size, void* d_ws, size_t ws_size,
                              hipStream_t stream) {
  (void)in_sizes; (void)n_in; (void)out_size; (void)ws_size;
  const float* x     = (const float*)d_in[0];   // [2,2048,1024]
  const float* Wqkv  = (const float*)d_in[1];   // [1024,3072]
  const float* bqkv  = (const float*)d_in[2];   // [3072]
  const float* Wproj = (const float*)d_in[3];   // [1024,1024]
  const float* bproj = (const float*)d_in[4];   // [1024]
  float* out = (float*)d_out;

  _Float16* xh     = (_Float16*)d_ws;                     // 4096*1024
  _Float16* wqkvT  = xh     + (size_t)4096 * 1024;        // 3072*1024 ([N][K])
  _Float16* wprojT = wqkvT  + (size_t)3072 * 1024;        // 1024*1024 ([N][K])
  _Float16* qh     = wprojT + (size_t)1024 * 1024;        // 32*2048*64
  _Float16* kh     = qh     + (size_t)32 * 2048 * 64;     // 32*2048*64
  _Float16* vth    = kh     + (size_t)32 * 2048 * 64;     // 32*64*2048 (transposed)
  _Float16* attnh  = vth    + (size_t)32 * 64 * 2048;     // 4096*1024

  cvt_f16 <<<16384, 256, 0, stream>>>(x, xh, 4096 * 1024);
  cvtT_f16<<<12288, 256, 0, stream>>>(Wqkv,  wqkvT,  3072, 1024);
  cvtT_f16<<< 4096, 256, 0, stream>>>(Wproj, wprojT, 1024, 1024);

  gemm_qkv <<<768, 256, 0, stream>>>(xh, wqkvT, bqkv, qh, kh, vth);
  attn_fwd <<<512, 256, 0, stream>>>(qh, kh, vth, attnh);
  gemm_proj<<<256, 256, 0, stream>>>(attnh, wprojT, bproj, out);
}
